// DFSGlimpseSingleObjectClassifier_72103910965835
// MI455X (gfx1250) — matrix operand order repaired and checked by reading, not compiled, not run on hardware
//
#include <hip/hip_runtime.h>
#include <hip/hip_bf16.h>
#include <math.h>

// ---------------------------------------------------------------------------
// DFS glimpse classifier for MI455X (gfx1250, wave32, WMMA).
// All heavy contractions run on v_wmma_f32_16x16x32_bf16 (fp32 accum).
// GEMMs are oriented so the big operand is the WMMA A-matrix: its wave32
// layout (lane=row, two contiguous 16B k-chunks) allows b128 loads from
// row-major memory. x is converted+transposed to bf16 once (13-step reuse).
// Conv im2col gathers run out of an LDS-staged activation slab.
// ---------------------------------------------------------------------------

typedef __attribute__((ext_vector_type(16))) __bf16 bf16x16;
typedef __attribute__((ext_vector_type(8)))  float  f32x8;

#define BATCH 128
#define CH    3
#define IMG   224   // H == W
#define DIM   128
#define PDIM  6
#define GG    15    // glimpse size
#define GP    16    // padded glimpse
#define NPIX  225   // 15*15

// ---------------- WMMA fragment gathers (ISA 7.12.2 wave32 layouts) --------

struct V32 { uint4 lo, hi; };

// A-matrix 16x32 bf16 from row-major source (lda elems, 16B-aligned rows).
// lane L: row m=L&15, k-half (L>>4)*8; frag elems map to row[khalf+0..7]
// and row[khalf+16..23] -> two aligned b128 loads.
__device__ __forceinline__ bf16x16 frag_a(const __bf16* __restrict__ src,
                                          int lda, int k0, int lane) {
  int m = lane & 15;
  int khalf = (lane >> 4) << 3;
  const __bf16* row = src + (size_t)m * lda + k0 + khalf;
  V32 t;
  t.lo = *(const uint4*)(row);
  t.hi = *(const uint4*)(row + 16);
  return __builtin_bit_cast(bf16x16, t);
}

// B-matrix 32x16 bf16, element B[k][n] = src[(k0+k)*ldb + n0 + n].
// lane L: col n=L&15; rows (L>>4)*8 + e (e<8), 16 + (L>>4)*8 + e-8 (e>=8).
__device__ __forceinline__ bf16x16 frag_b(const __bf16* __restrict__ src,
                                          int ldb, int k0, int n0, int lane) {
  int n = n0 + (lane & 15);
  int kb = k0 + ((lane >> 4) << 3);
  bf16x16 b;
#pragma unroll
  for (int e = 0; e < 16; ++e)
    b[e] = src[(size_t)(kb + e + ((e >> 3) << 3)) * ldb + n];
  return b;
}

// im2col B gather from LDS-staged activations: B[k][n], k=cin*9+dy*3+dx.
__device__ __forceinline__ bf16x16 frag_b_im2col(const __bf16* act /*LDS*/,
                                                 int Kvalid, int k0, int n0,
                                                 int lane) {
  int n = n0 + (lane & 15);
  bool nok = (n < NPIX);
  int y = n / GG, x = n - y * GG;
  int kb = k0 + ((lane >> 4) << 3);
  bf16x16 b;
#pragma unroll
  for (int e = 0; e < 16; ++e) {
    int k = kb + e + ((e >> 3) << 3);
    __bf16 v = (__bf16)0.0f;
    if (nok && k < Kvalid) {
      int cin = k / 9;
      int r = k - cin * 9;
      int dy = r / 3, dx = r - dy * 3;
      int iy = y + dy - 1, ix = x + dx - 1;
      if (iy >= 0 && iy < GG && ix >= 0 && ix < GG)
        v = act[cin * NPIX + iy * GG + ix];
    }
    b[e] = v;
  }
  return b;
}

__device__ __forceinline__ f32x8 wmma_bf16(bf16x16 a, bf16x16 b, f32x8 c) {
  return __builtin_amdgcn_wmma_f32_16x16x32_bf16(false, a, false, b,
                                                 (short)0, c, false, false);
}

// ---------------- utility kernels ------------------------------------------

__global__ void k_zero_u32(uint32_t* p, int n) {
  int i = blockIdx.x * blockDim.x + threadIdx.x;
  if (i < n) p[i] = 0u;
}

// x [bc][h][w] f32 -> xT [bc][w][h] bf16, 16x16 LDS-tiled transpose
__global__ void k_xpose(const float* __restrict__ x, __bf16* __restrict__ xT) {
  __shared__ float tile[16][17];
  int bc = blockIdx.z;
  int h0 = blockIdx.y * 16, w0 = blockIdx.x * 16;
  int tx = threadIdx.x, ty = threadIdx.y;
  const float* src = x + (size_t)bc * IMG * IMG;
  __bf16* dst = xT + (size_t)bc * IMG * IMG;
  tile[ty][tx] = src[(h0 + ty) * IMG + w0 + tx];
  __syncthreads();
  dst[(size_t)(w0 + ty) * IMG + h0 + tx] = (__bf16)tile[tx][ty];
}

// conv weights (Cout, Cin*9) f32 -> bf16 zero-padded to (Cout, Kpad)
__global__ void k_cvt_w(const float* __restrict__ w, __bf16* __restrict__ dst,
                        int M, int K, int Kpad) {
  int i = blockIdx.x * blockDim.x + threadIdx.x;
  if (i >= M * Kpad) return;
  int m = i / Kpad, k = i - m * Kpad;
  dst[i] = (k < K) ? (__bf16)w[m * K + k] : (__bf16)0.0f;
}

// ---------------- per-step small linears + glimpse params ------------------
__global__ void k_linear(const float* __restrict__ h_i,
                         const float* __restrict__ h_u,   // nullptr => msg 0
                         float* __restrict__ bx_i, float* __restrict__ a_i,
                         const float* __restrict__ Wb, const float* __restrict__ bb,
                         const float* __restrict__ Wa, const float* __restrict__ ba,
                         float* __restrict__ params) {
  __shared__ float h1[DIM];
  __shared__ float outs[8];
  int b = blockIdx.x, t = threadIdx.x;
  for (int d = t; d < DIM; d += 32)
    h1[d] = h_i[b * DIM + d] + (h_u ? h_u[b * DIM + d] : 0.0f);
  __syncthreads();
  if (t < 7) {
    float s = 0.0f;
    if (t < 6) {
      for (int d = 0; d < DIM; ++d) s += h1[d] * Wb[d * PDIM + t];
      outs[t] = bx_i[b * PDIM + t] + s + bb[t];
    } else {
      for (int d = 0; d < DIM; ++d) s += h1[d] * Wa[d];
      outs[6] = s + ba[0];
    }
  }
  __syncthreads();
  if (t < 6) bx_i[b * PDIM + t] = outs[t];
  if (t == 6) a_i[b] = outs[6];
  if (t == 0) {
    float cx = (tanhf(outs[0]) + 1.0f) * 0.5f * (IMG - 1);
    float cy = (tanhf(outs[1]) + 1.0f) * 0.5f * (IMG - 1);
    float dx = (1.0f / (1.0f + expf(-outs[2]))) * (float)(IMG - 1) / (GG - 1);
    float dy = (1.0f / (1.0f + expf(-outs[3]))) * (float)(IMG - 1) / (GG - 1);
    float sx = expf(outs[4]) * dx + 1e-3f;
    float sy = expf(outs[5]) * dy + 1e-3f;
    float* pp = params + b * 6;
    pp[0] = cx; pp[1] = cy; pp[2] = dx; pp[3] = dy; pp[4] = sx; pp[5] = sy;
  }
}

// ---------------- gaussian filter banks ------------------------------------
// FyT[b][h][g16]  (B operand of stage A; col g=15 stays zero)
// FxR[b][g16][w]  (A operand of stage B; row g=15 stays zero)
__global__ void k_filters(const float* __restrict__ params,
                          __bf16* __restrict__ FyT, __bf16* __restrict__ FxR) {
  __shared__ float vals[IMG];
  __shared__ float red[64];
  int blk = blockIdx.x, t = threadIdx.x;
  int g = blk % GG;
  int axis = (blk / GG) & 1;  // 0: Fy (cy,dy,sy)   1: Fx (cx,dx,sx)
  int b = blk / (2 * GG);
  const float* pp = params + b * 6;
  float c  = axis ? pp[0] : pp[1];
  float dd = axis ? pp[2] : pp[3];
  float s  = axis ? pp[4] : pp[5];
  float mu = c + ((float)g - (GG - 1) * 0.5f) * dd;
  float part = 0.0f;
  for (int i = t; i < IMG; i += 64) {
    float z = ((float)i - mu) / s;
    float e = expf(-0.5f * z * z);
    vals[i] = e;
    part += e;
  }
  red[t] = part;
  __syncthreads();
  for (int o = 32; o > 0; o >>= 1) {
    if (t < o) red[t] += red[t + o];
    __syncthreads();
  }
  float inv = 1.0f / (red[0] + 1e-8f);
  if (axis == 0) {
    __bf16* dst = FyT + (size_t)b * IMG * GP;
    for (int i = t; i < IMG; i += 64) dst[i * GP + g] = (__bf16)(vals[i] * inv);
  } else {
    __bf16* dst = FxR + ((size_t)b * GP + g) * IMG;
    for (int i = t; i < IMG; i += 64) dst[i] = (__bf16)(vals[i] * inv);
  }
}

// ---------------- glimpse stage A: tT[b,c] = xT[b,c] (224x224) @ FyT[b] (224x16)
__global__ void k_glimpseA(const __bf16* __restrict__ xT,
                           const __bf16* __restrict__ FyT,
                           __bf16* __restrict__ tT) {
  int bc = blockIdx.x, lane = threadIdx.x;
  int b = bc / CH;
  const __bf16* A  = xT + (size_t)bc * IMG * IMG;
  const __bf16* Bm = FyT + (size_t)b * IMG * GP;
  __bf16* T = tT + (size_t)bc * IMG * GP;
  bf16x16 bfr[7];
#pragma unroll
  for (int ks = 0; ks < 7; ++ks) bfr[ks] = frag_b(Bm, GP, ks * 32, 0, lane);
  int nl = lane & 15, mb = (lane >> 4) << 3;
  for (int mt = 0; mt < 14; ++mt) {
    const __bf16* Atile = A + (size_t)mt * 16 * IMG;
    if (mt + 1 < 14)
      __builtin_prefetch(Atile + 16 * IMG + (size_t)lane * IMG, 0, 0);
    f32x8 acc = {};
#pragma unroll
    for (int ks = 0; ks < 7; ++ks)
      acc = wmma_bf16(frag_a(Atile, IMG, ks * 32, lane), bfr[ks], acc);
#pragma unroll
    for (int r = 0; r < 8; ++r)
      T[(size_t)(mt * 16 + mb + r) * GP + nl] = (__bf16)acc[r];
  }
}

// ---------------- glimpse stage B: gT[b,c] = FxR[b] (16x224) @ tT[b,c] (224x16)
__global__ void k_glimpseB(const __bf16* __restrict__ FxR,
                           const __bf16* __restrict__ tT,
                           __bf16* __restrict__ gbuf) {
  int bc = blockIdx.x, lane = threadIdx.x;
  int b = bc / CH;
  const __bf16* A  = FxR + (size_t)b * GP * IMG;
  const __bf16* Bm = tT + (size_t)bc * IMG * GP;
  f32x8 acc = {};
#pragma unroll
  for (int ks = 0; ks < 7; ++ks)
    acc = wmma_bf16(frag_a(A, IMG, ks * 32, lane),
                    frag_b(Bm, GP, ks * 32, 0, lane), acc);
  int nl = lane & 15, mb = (lane >> 4) << 3;   // m = gx, n = gy
  __bf16* G = gbuf + (size_t)bc * NPIX;
  if (nl < GG) {
#pragma unroll
    for (int r = 0; r < 8; ++r) {
      int m = mb + r;
      if (m < GG) G[nl * GG + m] = (__bf16)acc[r];   // g[gy][gx] = gT[gx][gy]
    }
  }
}

// ---------------- conv layer as implicit GEMM (LDS-staged activations) -----
__global__ void k_conv(const __bf16* __restrict__ actIn,
                       __bf16* __restrict__ outBf, float* __restrict__ outF32,
                       const __bf16* __restrict__ wA,
                       const float* __restrict__ bias,
                       int Cin, int Cout, int Kpad, int leaky) {
  __shared__ __bf16 sact[64 * NPIX];           // up to 28.8 KB
  int blk = blockIdx.x, lane = threadIdx.x;
  int b = blk / 15, nt = blk % 15;
  const __bf16* act = actIn + (size_t)b * Cin * NPIX;
  int total = Cin * NPIX;
  if (Cin >= 16) {                             // slab bytes multiple of 16
    const uint4* src = (const uint4*)act;
    uint4* dst = (uint4*)sact;
    int nv = total >> 3;
    for (int i = lane; i < nv; i += 32) dst[i] = src[i];
  } else {
    for (int i = lane; i < total; i += 32) sact[i] = act[i];
  }
  __syncthreads();

  int MT = Cout >> 4;
  int Kvalid = Cin * 9;
  f32x8 zero = {};
  f32x8 acc[4];
#pragma unroll
  for (int mt = 0; mt < 4; ++mt) acc[mt] = zero;
  int nsteps = Kpad >> 5;
  for (int ks = 0; ks < nsteps; ++ks) {
    bf16x16 bfrag = frag_b_im2col(sact, Kvalid, ks * 32, nt * 16, lane);
    for (int mt = 0; mt < MT; ++mt) {
      bf16x16 afrag = frag_a(wA + (size_t)mt * 16 * Kpad, Kpad, ks * 32, lane);
      acc[mt] = wmma_bf16(afrag, bfrag, acc[mt]);
    }
  }
  int nl = lane & 15, mb = (lane >> 4) << 3;
  int n = nt * 16 + nl;
  if (n < NPIX) {
    for (int mt = 0; mt < MT; ++mt) {
#pragma unroll
      for (int r = 0; r < 8; ++r) {
        int m = mt * 16 + mb + r;
        float v = acc[mt][r] + bias[m];
        if (leaky) v = (v > 0.0f) ? v : 0.01f * v;
        size_t o = ((size_t)b * Cout + m) * NPIX + n;
        if (outF32) outF32[o] = v;
        else        outBf[o]  = (__bf16)v;
      }
    }
  }
}

// ---------------- adaptive maxpool 2x2 + flatten + h accumulate ------------
__global__ void k_pool_accum(const float* __restrict__ act4,
                             float* __restrict__ h_i) {
  int b = blockIdx.x, d = threadIdx.x;  // d = c*4 + py*2 + px
  int c = d >> 2, q = d & 3, py = q >> 1, px = q & 1;
  const float* base = act4 + ((size_t)b * 32 + c) * NPIX;
  int y0 = py ? 7 : 0, x0 = px ? 7 : 0;
  float m = -3.4e38f;
  for (int yy = 0; yy < 8; ++yy)
    for (int xx = 0; xx < 8; ++xx)
      m = fmaxf(m, base[(y0 + yy) * GG + x0 + xx]);
  h_i[b * DIM + d] += m;
}

// ---------------- readout: relu((a0 * h0) @ Wr + br) -----------------------
__global__ void k_readout(const float* __restrict__ h0,
                          const float* __restrict__ a0,
                          const float* __restrict__ Wr,
                          const float* __restrict__ br,
                          float* __restrict__ out) {
  __shared__ float hv[DIM];
  int b = blockIdx.x, t = threadIdx.x;
  float s = a0[b];
  for (int d = t; d < DIM; d += 32) hv[d] = h0[b * DIM + d] * s;
  __syncthreads();
  if (t < 10) {
    float acc = 0.0f;
    for (int d = 0; d < DIM; ++d) acc += hv[d] * Wr[d * 10 + t];
    acc += br[t];
    out[b * 10 + t] = fmaxf(acc, 0.0f);
  }
}

// ---------------------------------------------------------------------------

extern "C" void kernel_launch(void* const* d_in, const int* in_sizes, int n_in,
                              void* d_out, int out_size, void* d_ws, size_t ws_size,
                              hipStream_t stream) {
  (void)in_sizes; (void)n_in; (void)out_size; (void)ws_size;
  // dict order with convs tuple flattened: x, Wb, bb, Wy, by, Wa, ba,
  // (w1,b1,w2,b2,w3,b3,w4,b4), Wr, br
  const float* x  = (const float*)d_in[0];
  const float* Wb = (const float*)d_in[1];
  const float* bb = (const float*)d_in[2];
  const float* Wa = (const float*)d_in[5];
  const float* ba = (const float*)d_in[6];
  const float* cw[4] = {(const float*)d_in[7],  (const float*)d_in[9],
                        (const float*)d_in[11], (const float*)d_in[13]};
  const float* cb[4] = {(const float*)d_in[8],  (const float*)d_in[10],
                        (const float*)d_in[12], (const float*)d_in[14]};
  const float* Wr = (const float*)d_in[15];
  const float* br = (const float*)d_in[16];
  float* out = (float*)d_out;

  // ---- workspace layout (state re-initialized every call => deterministic)
  char* ws = (char*)d_ws;
  auto al = [](size_t v) { return (v + 255) & ~(size_t)255; };
  size_t off = 0;
  const size_t ZB = off;                                    // zero-region start
  float* hbuf  = (float*)(ws + off); off += 7 * BATCH * DIM * 4;
  float* bxbuf = (float*)(ws + off); off += 7 * BATCH * PDIM * 4;
  float* abuf  = (float*)(ws + off); off += 7 * BATCH * 4;
  __bf16* FyT  = (__bf16*)(ws + off); off += (size_t)BATCH * IMG * GP * 2;
  __bf16* FxR  = (__bf16*)(ws + off); off += (size_t)BATCH * GP * IMG * 2;
  const size_t ZEROB = off - ZB;
  off = al(off);
  float*  params = (float*)(ws + off); off = al(off + BATCH * 6 * 4);
  __bf16* xT   = (__bf16*)(ws + off); off = al(off + (size_t)BATCH * CH * IMG * IMG * 2);
  __bf16* tT   = (__bf16*)(ws + off); off = al(off + (size_t)BATCH * CH * IMG * GP * 2);
  __bf16* gbuf = (__bf16*)(ws + off); off = al(off + (size_t)BATCH * CH * NPIX * 2);
  __bf16* act1 = (__bf16*)(ws + off); off = al(off + (size_t)BATCH * 16 * NPIX * 2);
  __bf16* act2 = (__bf16*)(ws + off); off = al(off + (size_t)BATCH * 32 * NPIX * 2);
  __bf16* act3 = (__bf16*)(ws + off); off = al(off + (size_t)BATCH * 64 * NPIX * 2);
  float*  act4 = (float*)(ws + off);  off = al(off + (size_t)BATCH * 32 * NPIX * 4);
  const int Cin[4]  = {3, 16, 32, 64};
  const int Cout[4] = {16, 32, 64, 32};
  const int Kpad[4] = {32, 160, 288, 576};
  __bf16* wbf[4];
  for (int l = 0; l < 4; ++l) { wbf[l] = (__bf16*)(ws + off); off = al(off + (size_t)Cout[l] * Kpad[l] * 2); }

  // ---- init: zero state, transpose+convert x, convert weights
  {
    int nz = (int)(ZEROB / 4);
    k_zero_u32<<<(nz + 255) / 256, 256, 0, stream>>>((uint32_t*)(ws + ZB), nz);
    k_xpose<<<dim3(IMG / 16, IMG / 16, BATCH * CH), dim3(16, 16), 0, stream>>>(x, xT);
    for (int l = 0; l < 4; ++l) {
      int n = Cout[l] * Kpad[l];
      k_cvt_w<<<(n + 255) / 256, 256, 0, stream>>>(cw[l], wbf[l], Cout[l], Cin[l] * 9, Kpad[l]);
    }
  }

  // ---- Euler-tour walk: update(0, 0) then update(v, h[u]) per edge
  static const int us[13] = {-1, 0, 1, 3, 1, 4, 1, 0, 2, 5, 2, 6, 2};
  static const int vs[13] = { 0, 1, 3, 1, 4, 1, 0, 2, 5, 2, 6, 2, 0};

  for (int step = 0; step < 13; ++step) {
    int v = vs[step], u = us[step];
    float* h_i  = hbuf + (size_t)v * BATCH * DIM;
    const float* h_u = (u >= 0) ? hbuf + (size_t)u * BATCH * DIM : nullptr;
    float* bx_i = bxbuf + (size_t)v * BATCH * PDIM;
    float* a_i  = abuf + (size_t)v * BATCH;

    k_linear<<<BATCH, 32, 0, stream>>>(h_i, h_u, bx_i, a_i, Wb, bb, Wa, ba, params);
    k_filters<<<BATCH * 2 * GG, 64, 0, stream>>>(params, FyT, FxR);
    k_glimpseA<<<BATCH * CH, 32, 0, stream>>>(xT, FyT, tT);
    k_glimpseB<<<BATCH * CH, 32, 0, stream>>>(FxR, tT, gbuf);
    k_conv<<<BATCH * 15, 32, 0, stream>>>(gbuf, act1, nullptr, wbf[0], cb[0], Cin[0], Cout[0], Kpad[0], 1);
    k_conv<<<BATCH * 15, 32, 0, stream>>>(act1, act2, nullptr, wbf[1], cb[1], Cin[1], Cout[1], Kpad[1], 1);
    k_conv<<<BATCH * 15, 32, 0, stream>>>(act2, act3, nullptr, wbf[2], cb[2], Cin[2], Cout[2], Kpad[2], 1);
    k_conv<<<BATCH * 15, 32, 0, stream>>>(act3, nullptr, act4, wbf[3], cb[3], Cin[3], Cout[3], Kpad[3], 0);
    k_pool_accum<<<BATCH, DIM, 0, stream>>>(act4, h_i);
  }

  k_readout<<<BATCH, 32, 0, stream>>>(hbuf, abuf, Wr, br, out);
}
